// MultiHeadAttention_79439715106973
// MI455X (gfx1250) — matrix operand order repaired and checked by reading, not compiled, not run on hardware
//
#include <hip/hip_runtime.h>

// ---------------------------------------------------------------------------
// MHA for MI455X (gfx1250): bf16 WMMA GEMMs + flash attention.
// B=4, S=2048, DIM=1024, H=16, Dh=64.
// ---------------------------------------------------------------------------

#define BATCH      4
#define SEQ        2048
#define DIM        1024
#define NHEAD      16
#define HDIM       64
#define MROWS      (BATCH * SEQ)      // 8192

typedef unsigned short u16;
typedef unsigned int   u32;
typedef __attribute__((ext_vector_type(16))) __bf16          v16bf;
typedef __attribute__((ext_vector_type(8)))  float           v8f;
typedef __attribute__((ext_vector_type(8)))  unsigned short  u16x8;

union Frag {               // one WMMA A or B operand (16 bf16 per lane)
    v16bf bf;
    u16x8 h[2];
    u16   us[16];
};

__device__ __forceinline__ u16 f2bf(float f) {
    union { float f; u32 u; } v; v.f = f;
    u32 r = v.u + 0x7FFFu + ((v.u >> 16) & 1u);   // round-to-nearest-even
    return (u16)(r >> 16);
}
__device__ __forceinline__ float bf2f(u16 h) {
    union { u32 u; float f; } v; v.u = ((u32)h) << 16;
    return v.f;
}

__device__ __forceinline__ v8f wmma_bf16(const Frag& a, const Frag& b, v8f c) {
    return __builtin_amdgcn_wmma_f32_16x16x32_bf16(
        /*neg_a=*/false, a.bf, /*neg_b=*/false, b.bf,
        /*c_mod=*/(short)0, c, /*reuse_a=*/false, /*reuse_b=*/false);
}

// ---------------------------------------------------------------------------
// 1) fp32 -> bf16 conversion
// ---------------------------------------------------------------------------
__global__ void cvt_f32_bf16(const float* __restrict__ in, u16* __restrict__ out, int n) {
    int i = blockIdx.x * blockDim.x + threadIdx.x;
    if (i < n) out[i] = f2bf(in[i]);
}

// ---------------------------------------------------------------------------
// 2) QKV projection: Y[m,e] = sum_d xbf[m,d] * wbf[e,d] + bias[e]
//    Output scattered into [B][H][S][HDIM] bf16.
//    Block = 256 thr (8 waves as 2Mx4N), block tile 64(M) x 128(N),
//    wave tile 32x32 = 4 WMMA C tiles, K step 32.
// ---------------------------------------------------------------------------
__global__ __launch_bounds__(256) void qkv_proj(
    const u16* __restrict__ xbf,
    const u16* __restrict__ qw, const u16* __restrict__ kw, const u16* __restrict__ vw,
    const float* __restrict__ qb, const float* __restrict__ kb, const float* __restrict__ vb,
    u16* __restrict__ Qo, u16* __restrict__ Ko, u16* __restrict__ Vo)
{
    const int lane = threadIdx.x & 31, wave = threadIdx.x >> 5;
    const int lm = lane & 15, lg = lane >> 4;
    const int mOff = blockIdx.x * 64  + (wave & 1) * 32;
    const int nOff = blockIdx.y * 128 + (wave >> 1) * 32;

    const u16* W; const float* bias; u16* out;
    if (blockIdx.z == 0)      { W = qw; bias = qb; out = Qo; }
    else if (blockIdx.z == 1) { W = kw; bias = kb; out = Ko; }
    else                      { W = vw; bias = vb; out = Vo; }

    v8f acc[2][2] = {{{0,0,0,0,0,0,0,0},{0,0,0,0,0,0,0,0}},
                     {{0,0,0,0,0,0,0,0},{0,0,0,0,0,0,0,0}}};

    for (int kbs = 0; kbs < DIM; kbs += 32) {
        Frag a[2], b[2];
#pragma unroll
        for (int t = 0; t < 2; t++) {
            const u16* ap = xbf + (size_t)(mOff + t*16 + lm) * DIM + kbs + lg*8;
            a[t].h[0] = *(const u16x8*)(ap);
            a[t].h[1] = *(const u16x8*)(ap + 16);
            const u16* bp = W + (size_t)(nOff + t*16 + lm) * DIM + kbs + lg*8;
            b[t].h[0] = *(const u16x8*)(bp);
            b[t].h[1] = *(const u16x8*)(bp + 16);
        }
#pragma unroll
        for (int mt = 0; mt < 2; mt++)
#pragma unroll
            for (int nt = 0; nt < 2; nt++)
                acc[mt][nt] = wmma_bf16(a[mt], b[nt], acc[mt][nt]);
    }

#pragma unroll
    for (int nt = 0; nt < 2; nt++) {
        const int e  = nOff + nt*16 + lm;
        const float bv = bias[e];
        const int h = e >> 6, d = e & (HDIM - 1);
#pragma unroll
        for (int mt = 0; mt < 2; mt++) {
#pragma unroll
            for (int r = 0; r < 8; r++) {
                const int m = mOff + mt*16 + r + lg*8;
                const int bb = m >> 11, s = m & (SEQ - 1);
                out[(((size_t)(bb * NHEAD + h)) * SEQ + s) * HDIM + d] =
                    f2bf(acc[mt][nt][r] + bv);
            }
        }
    }
}

// ---------------------------------------------------------------------------
// 3) RoPE applied in place to bf16 Q and K ([B*H][S][64], pairs (i, i+32))
// ---------------------------------------------------------------------------
__global__ void rope_kernel(u16* __restrict__ Q, u16* __restrict__ K) {
    const int idx = blockIdx.x * blockDim.x + threadIdx.x;
    if (idx >= BATCH * NHEAD * SEQ * 32) return;
    const int i  = idx & 31;
    const int s  = (idx >> 5) & (SEQ - 1);
    const int bh = idx >> 16;
    // freq = 10000^(-i/32)  (log(10000) = 9.210340371976184)
    const float freq = __expf(-(float)i * (9.210340371976184f / 32.0f));
    float sn, cs;
    __sincosf((float)s * freq, &sn, &cs);
    const size_t base = ((size_t)bh * SEQ + s) * HDIM;

    float ql = bf2f(Q[base + i]), qr = bf2f(Q[base + i + 32]);
    Q[base + i]      = f2bf(ql * cs - qr * sn);
    Q[base + i + 32] = f2bf(ql * sn + qr * cs);

    float kl = bf2f(K[base + i]), kr = bf2f(K[base + i + 32]);
    K[base + i]      = f2bf(kl * cs - kr * sn);
    K[base + i + 32] = f2bf(kl * sn + kr * cs);
}

// ---------------------------------------------------------------------------
// 4) Flash attention. Block = 256 thr (8 waves), one (b,h) per blockIdx.y,
//    128 query rows per block (16 per wave). Keys streamed in blocks of 32:
//      scores  : 2 key tiles x 2 K-dim chunks = 4 WMMAs
//      P @ V   : 4 d tiles  = 4 WMMAs
//    K tile staged row-major in LDS (stride 72 halves), V tile staged
//    transposed (stride 40), P re-laid out via per-wave LDS scratch.
// ---------------------------------------------------------------------------
#define KSTR 72
#define VSTR 40
#define PSTR 40

__global__ __launch_bounds__(256) void attention_kernel(
    const u16* __restrict__ Q, const u16* __restrict__ K, const u16* __restrict__ V,
    const unsigned char* __restrict__ mask, u16* __restrict__ aout)
{
    __shared__ u16 ldsK[32 * KSTR];
    __shared__ u16 ldsV[HDIM * VSTR];
    __shared__ u16 ldsP[8 * 16 * PSTR];

    const int lane = threadIdx.x & 31, wave = threadIdx.x >> 5;
    const int lm = lane & 15, lg = lane >> 4;
    const int bh = blockIdx.y, b = bh >> 4, hh = bh & (NHEAD - 1);
    const size_t headBase = (size_t)bh * SEQ * HDIM;
    const int q0 = blockIdx.x * 128 + wave * 16;

    // Q fragments for this wave's 16 rows (d 0..31 and 32..63), reused all loop
    Frag aQ[2];
    {
        const u16* qp = Q + headBase + (size_t)(q0 + lm) * HDIM;
        aQ[0].h[0] = *(const u16x8*)(qp + lg*8);
        aQ[0].h[1] = *(const u16x8*)(qp + lg*8 + 16);
        aQ[1].h[0] = *(const u16x8*)(qp + 32 + lg*8);
        aQ[1].h[1] = *(const u16x8*)(qp + 32 + lg*8 + 16);
    }

    v8f o[4] = {{0,0,0,0,0,0,0,0},{0,0,0,0,0,0,0,0},
                {0,0,0,0,0,0,0,0},{0,0,0,0,0,0,0,0}};
    float mrow[8], lrow[8];
#pragma unroll
    for (int r = 0; r < 8; r++) { mrow[r] = -1e30f; lrow[r] = 0.0f; }

    const float scale = 0.125f;           // 1/sqrt(64)
    u16* myP = ldsP + wave * 16 * PSTR;

    for (int kbs = 0; kbs < SEQ; kbs += 32) {
        // ---- cooperative staging: 32 keys x 64 d, 8 elems per thread ----
        {
            const int t8  = threadIdx.x * 8;
            const int key = t8 >> 6;
            const int d0  = t8 & (HDIM - 1);
            const u16x8 kv = *(const u16x8*)(K + headBase + (size_t)(kbs + key) * HDIM + d0);
            *(u16x8*)(&ldsK[key * KSTR + d0]) = kv;                 // row-major
            const u16x8 vv = *(const u16x8*)(V + headBase + (size_t)(kbs + key) * HDIM + d0);
#pragma unroll
            for (int j = 0; j < 8; j++)                             // transposed
                ldsV[(d0 + j) * VSTR + key] = vv[j];
        }
        __syncthreads();

        // ---- scores: 2 key tiles of 16, each accumulates over d in 2 WMMAs ----
        v8f sc[2];
#pragma unroll
        for (int t = 0; t < 2; t++) {
            v8f s = {0,0,0,0,0,0,0,0};
#pragma unroll
            for (int f = 0; f < 2; f++) {
                Frag bK;
                const u16* p = &ldsK[(t*16 + lm) * KSTR + f*32 + lg*8];
                bK.h[0] = *(const u16x8*)(p);
                bK.h[1] = *(const u16x8*)(p + 16);
                s = wmma_bf16(aQ[f], bK, s);
            }
            const int keyAbs = kbs + t*16 + lm;
            const float madd = mask[b * SEQ + keyAbs] ? -1.0e9f : 0.0f;
#pragma unroll
            for (int r = 0; r < 8; r++) s[r] = s[r] * scale + madd;
            sc[t] = s;
        }

        // ---- online softmax per row (rows striped: VGPR r -> M = r + 8*lg) ----
#pragma unroll
        for (int r = 0; r < 8; r++) {
            float v0 = sc[0][r], v1 = sc[1][r];
            float mx = fmaxf(v0, v1);
            mx = fmaxf(mx, __shfl_xor(mx, 1, 32));
            mx = fmaxf(mx, __shfl_xor(mx, 2, 32));
            mx = fmaxf(mx, __shfl_xor(mx, 4, 32));
            mx = fmaxf(mx, __shfl_xor(mx, 8, 32));
            const float mnew  = fmaxf(mrow[r], mx);
            const float alpha = __expf(mrow[r] - mnew);
            const float p0 = __expf(v0 - mnew);
            const float p1 = __expf(v1 - mnew);
            float rs = p0 + p1;
            rs += __shfl_xor(rs, 1, 32);
            rs += __shfl_xor(rs, 2, 32);
            rs += __shfl_xor(rs, 4, 32);
            rs += __shfl_xor(rs, 8, 32);
            lrow[r] = lrow[r] * alpha + rs;
            mrow[r] = mnew;
#pragma unroll
            for (int dt = 0; dt < 4; dt++) o[dt][r] *= alpha;
            const int M = r + 8 * lg;
            myP[M * PSTR + lm]      = f2bf(p0);   // keys kbs+0..15
            myP[M * PSTR + 16 + lm] = f2bf(p1);   // keys kbs+16..31
        }

        // ---- P (16x32) as A fragment, then P @ V over 4 d tiles ----
        Frag aP;
        aP.h[0] = *(const u16x8*)(&myP[lm * PSTR + lg*8]);
        aP.h[1] = *(const u16x8*)(&myP[lm * PSTR + 16 + lg*8]);
#pragma unroll
        for (int dt = 0; dt < 4; dt++) {
            Frag bV;
            const u16* p = &ldsV[(dt*16 + lm) * VSTR + lg*8];
            bV.h[0] = *(const u16x8*)(p);
            bV.h[1] = *(const u16x8*)(p + 16);
            o[dt] = wmma_bf16(aP, bV, o[dt]);
        }
        __syncthreads();
    }

    // ---- epilogue: normalize, write [B][S][DIM] bf16 (col = h*64 + d) ----
#pragma unroll
    for (int dt = 0; dt < 4; dt++) {
#pragma unroll
        for (int r = 0; r < 8; r++) {
            const int qrow = q0 + r + 8*lg;
            const float val = o[dt][r] / lrow[r];
            aout[(size_t)(b * SEQ + qrow) * DIM + hh * HDIM + dt*16 + lm] = f2bf(val);
        }
    }
}

// ---------------------------------------------------------------------------
// 5) Output projection: out[m,e] = sum_d aout[m,d]*ow[e,d] + ob[e]  (fp32)
// ---------------------------------------------------------------------------
__global__ __launch_bounds__(256) void out_proj(
    const u16* __restrict__ abf, const u16* __restrict__ ow,
    const float* __restrict__ ob, float* __restrict__ out)
{
    const int lane = threadIdx.x & 31, wave = threadIdx.x >> 5;
    const int lm = lane & 15, lg = lane >> 4;
    const int mOff = blockIdx.x * 64  + (wave & 1) * 32;
    const int nOff = blockIdx.y * 128 + (wave >> 1) * 32;

    v8f acc[2][2] = {{{0,0,0,0,0,0,0,0},{0,0,0,0,0,0,0,0}},
                     {{0,0,0,0,0,0,0,0},{0,0,0,0,0,0,0,0}}};

    for (int kbs = 0; kbs < DIM; kbs += 32) {
        Frag a[2], b[2];
#pragma unroll
        for (int t = 0; t < 2; t++) {
            const u16* ap = abf + (size_t)(mOff + t*16 + lm) * DIM + kbs + lg*8;
            a[t].h[0] = *(const u16x8*)(ap);
            a[t].h[1] = *(const u16x8*)(ap + 16);
            const u16* bp = ow + (size_t)(nOff + t*16 + lm) * DIM + kbs + lg*8;
            b[t].h[0] = *(const u16x8*)(bp);
            b[t].h[1] = *(const u16x8*)(bp + 16);
        }
#pragma unroll
        for (int mt = 0; mt < 2; mt++)
#pragma unroll
            for (int nt = 0; nt < 2; nt++)
                acc[mt][nt] = wmma_bf16(a[mt], b[nt], acc[mt][nt]);
    }

#pragma unroll
    for (int nt = 0; nt < 2; nt++) {
        const int e = nOff + nt*16 + lm;
        const float bv = ob[e];
#pragma unroll
        for (int mt = 0; mt < 2; mt++) {
#pragma unroll
            for (int r = 0; r < 8; r++) {
                const int m = mOff + mt*16 + r + lg*8;
                out[(size_t)m * DIM + e] = acc[mt][nt][r] + bv;
            }
        }
    }
}

// ---------------------------------------------------------------------------
// Launch
// ---------------------------------------------------------------------------
extern "C" void kernel_launch(void* const* d_in, const int* in_sizes, int n_in,
                              void* d_out, int out_size, void* d_ws, size_t ws_size,
                              hipStream_t stream) {
    const float* x    = (const float*)d_in[0];
    const unsigned char* mask = (const unsigned char*)d_in[1];
    const float* q_w  = (const float*)d_in[2];
    const float* q_b  = (const float*)d_in[3];
    const float* k_w  = (const float*)d_in[4];
    const float* k_b  = (const float*)d_in[5];
    const float* v_w  = (const float*)d_in[6];
    const float* v_b  = (const float*)d_in[7];
    const float* o_w  = (const float*)d_in[8];
    const float* o_b  = (const float*)d_in[9];
    float* out = (float*)d_out;

    // workspace carve-up (all bf16 element counts)
    char* ws = (char*)d_ws;
    size_t off = 0;
    u16* xbf  = (u16*)(ws + off); off += (size_t)MROWS * DIM * 2;          // 16.8 MB
    u16* qwbf = (u16*)(ws + off); off += (size_t)DIM * DIM * 2;
    u16* kwbf = (u16*)(ws + off); off += (size_t)DIM * DIM * 2;
    u16* vwbf = (u16*)(ws + off); off += (size_t)DIM * DIM * 2;
    u16* owbf = (u16*)(ws + off); off += (size_t)DIM * DIM * 2;            // +8.4 MB
    u16* Qh   = (u16*)(ws + off); off += (size_t)MROWS * DIM * 2;
    u16* Kh   = (u16*)(ws + off); off += (size_t)MROWS * DIM * 2;
    u16* Vh   = (u16*)(ws + off); off += (size_t)MROWS * DIM * 2;          // +50.3 MB
    u16* abf  = (u16*)(ws + off); off += (size_t)MROWS * DIM * 2;          // +16.8 MB

    const int nX = MROWS * DIM;        // 8,388,608
    const int nW = DIM * DIM;          // 1,048,576
    cvt_f32_bf16<<<(nX + 255) / 256, 256, 0, stream>>>(x,   xbf,  nX);
    cvt_f32_bf16<<<(nW + 255) / 256, 256, 0, stream>>>(q_w, qwbf, nW);
    cvt_f32_bf16<<<(nW + 255) / 256, 256, 0, stream>>>(k_w, kwbf, nW);
    cvt_f32_bf16<<<(nW + 255) / 256, 256, 0, stream>>>(v_w, vwbf, nW);
    cvt_f32_bf16<<<(nW + 255) / 256, 256, 0, stream>>>(o_w, owbf, nW);

    qkv_proj<<<dim3(MROWS / 64, DIM / 128, 3), 256, 0, stream>>>(
        xbf, qwbf, kwbf, vwbf, q_b, k_b, v_b, Qh, Kh, Vh);

    const int nR = BATCH * NHEAD * SEQ * 32;   // 4,194,304
    rope_kernel<<<(nR + 255) / 256, 256, 0, stream>>>(Qh, Kh);

    attention_kernel<<<dim3(SEQ / 128, BATCH * NHEAD), 256, 0, stream>>>(
        Qh, Kh, Vh, mask, abf);

    out_proj<<<dim3(MROWS / 64, DIM / 128), 256, 0, stream>>>(abf, owbf, o_b, out);
}
